// GraphTokenizer_65292092834112
// MI455X (gfx1250) — matrix operand order repaired and checked by reading, not compiled, not run on hardware
//
#include <hip/hip_runtime.h>
#include <hip/hip_bf16.h>
#include <stdint.h>

// Problem constants (fixed by the reference).
#define N_NODES 65536
#define DIMS    512
#define MCODE   4096

// Tiling.
#define BLOCK_ROWS 64      // embedding rows per workgroup (4 waves x 16)
#define THREADS    128
#define MBLOCK     128     // codebook columns per M-block (8 x 16 tiles)
#define KCHUNK     32      // K depth per WMMA
#define NKC        (DIMS / KCHUNK)   // 16 K-chunks

// LDS layouts (packed bf16, 2 per dword). Strides padded, 16B-aligned.
#define A_STRIDE 260       // dwords per A row (256 data + 4 pad)
#define B_STRIDE 20        // dwords per B row (16 data + 4 pad)

typedef __attribute__((ext_vector_type(16))) __bf16 v16bf;
typedef __attribute__((ext_vector_type(2)))  __bf16 v2bf;
typedef __attribute__((ext_vector_type(8)))  float  v8f;

struct BF32 { uint4 lo; uint4 hi; };   // 32B = one v16bf fragment

// Pack two f32 into one dword of bf16x2 -> exactly one v_cvt_pk_bf16_f32.
static __device__ inline uint32_t pack_bf16(float a, float b) {
#if __has_builtin(__builtin_amdgcn_cvt_pk_bf16_f32)
    return __builtin_bit_cast(uint32_t, __builtin_amdgcn_cvt_pk_bf16_f32(a, b));
#else
    v2bf p;
    p.x = (__bf16)a;
    p.y = (__bf16)b;
    return __builtin_bit_cast(uint32_t, p);
#endif
}

// ---------------------------------------------------------------------------
// c_sq[m] = ||codebook[m]||^2   (one wave per row)
// ---------------------------------------------------------------------------
__global__ __launch_bounds__(256)
void csq_kernel(const float* __restrict__ cb, float* __restrict__ csq) {
    const int lane = threadIdx.x & 31;
    const int wave = threadIdx.x >> 5;
    const int m = blockIdx.x * 8 + wave;
    const float* row = cb + (size_t)m * DIMS;
    float s = 0.0f;
#pragma unroll
    for (int j = 0; j < DIMS / 32; ++j) {
        float v = row[lane + 32 * j];
        s = __builtin_fmaf(v, v, s);
    }
#pragma unroll
    for (int msk = 16; msk >= 1; msk >>= 1)
        s += __shfl_xor(s, msk, 32);
    if (lane == 0) csq[m] = s;
}

// ---------------------------------------------------------------------------
// Fused: cross-GEMM (bf16 WMMA, f32 acc) + running argmin + gather epilogue.
// A tile staged once in LDS as bf16; B chunks double-buffered bf16.
// ---------------------------------------------------------------------------
__global__ __launch_bounds__(THREADS)
void assign_kernel(const float* __restrict__ emb,
                   const float* __restrict__ cb,
                   const float* __restrict__ csq,
                   float* __restrict__ outQ,
                   float* __restrict__ outIdx) {
    __shared__ uint32_t sA[BLOCK_ROWS * A_STRIDE];   // 65.0 KB: full-K A tile, bf16
    __shared__ uint32_t sB[2][MBLOCK * B_STRIDE];    // 20.0 KB: double-buffered B

    const int tid      = threadIdx.x;
    const int lane     = tid & 31;
    const int wave     = tid >> 5;
    const int blockRow = blockIdx.x * BLOCK_ROWS;

    const int halfSel = lane >> 4;   // 0: lanes 0-15, 1: lanes 16-31
    const int l16     = lane & 15;

    // -------- Stage A (64 rows x 512 f32 -> bf16) once --------------------
    // 2048 contiguous 16-f32 segments; 16 per thread.
#pragma unroll
    for (int i = 0; i < 16; ++i) {
        const int s = tid + THREADS * i;
        const int r = s >> 5;            // row 0..63
        const int c = (s & 31) * 16;     // f32 column
        const float4* src = (const float4*)(emb + (size_t)(blockRow + r) * DIMS + c);
        const float4 f0 = src[0], f1 = src[1], f2 = src[2], f3 = src[3];
        uint32_t* d = sA + r * A_STRIDE + (c >> 1);
        uint4 p0 = { pack_bf16(f0.x, f0.y), pack_bf16(f0.z, f0.w),
                     pack_bf16(f1.x, f1.y), pack_bf16(f1.z, f1.w) };
        uint4 p1 = { pack_bf16(f2.x, f2.y), pack_bf16(f2.z, f2.w),
                     pack_bf16(f3.x, f3.y), pack_bf16(f3.z, f3.w) };
        ((uint4*)d)[0] = p0;
        ((uint4*)d)[1] = p1;
    }
    __syncthreads();

    // Running per-lane best distance / index.
    // C-layout: slot i = row M=i (lanes 0-15) or M=i+8 (lanes 16-31),
    // column N = mBase + t*16 + l16.
    float best[8];
    int   bidx[8];
#pragma unroll
    for (int i = 0; i < 8; ++i) { best[i] = 3.402823466e+38f; bidx[i] = 0; }

    for (int mb = 0; mb < MCODE / MBLOCK; ++mb) {
        const int mBase = mb * MBLOCK;
        const float* brow = cb + (size_t)(mBase + tid) * DIMS;  // my codebook row

        v8f acc[8];
#pragma unroll
        for (int t = 0; t < 8; ++t) acc[t] = (v8f){0,0,0,0,0,0,0,0};

        // Prefetch K-chunk 0 of my B row into registers.
        float4 pre[8];
        {
            const float4* g = (const float4*)brow;
#pragma unroll
            for (int j = 0; j < 8; ++j) pre[j] = g[j];
        }

        for (int kc = 0; kc < NKC; ++kc) {
            const int buf = kc & 1;

            // Store prefetched chunk (convert f32->bf16 once) into sB[buf].
            // Safe without a pre-barrier: last readers of sB[buf] ran before
            // the barrier of iteration kc-1.
            {
                uint32_t* d = sB[buf] + tid * B_STRIDE;
#pragma unroll
                for (int q = 0; q < 4; ++q) {
                    const float4 a = pre[2 * q], b = pre[2 * q + 1];
                    uint4 p = { pack_bf16(a.x, a.y), pack_bf16(a.z, a.w),
                                pack_bf16(b.x, b.y), pack_bf16(b.z, b.w) };
                    ((uint4*)d)[q] = p;
                }
            }
            // Issue next chunk's global loads before the barrier (overlap).
            if (kc + 1 < NKC) {
                const float4* g = (const float4*)(brow + (kc + 1) * KCHUNK);
#pragma unroll
                for (int j = 0; j < 8; ++j) pre[j] = g[j];
            }
            __syncthreads();

            // A fragment: 2 x ds_load_b128, no conversion.
            BF32 araw;
            {
                const uint32_t* ap =
                    sA + (wave * 16 + l16) * A_STRIDE + kc * 16 + halfSel * 4;
                araw.lo = ((const uint4*)ap)[0];        // K k0+0..7  / k0+8..15
                araw.hi = ((const uint4*)(ap + 8))[0];  // K k0+16..23/ k0+24..31
            }
            const v16bf aF = __builtin_bit_cast(v16bf, araw);

            // 8 column tiles: 2 x ds_load_b128 + 1 WMMA each.
#pragma unroll
            for (int t = 0; t < 8; ++t) {
                const uint32_t* bp =
                    sB[buf] + (t * 16 + l16) * B_STRIDE + halfSel * 4;
                BF32 braw;
                braw.lo = ((const uint4*)bp)[0];   // K 0..7   / 8..15
                braw.hi = ((const uint4*)bp)[2];   // K 16..23 / 24..31
                const v16bf bF = __builtin_bit_cast(v16bf, braw);
                acc[t] = __builtin_amdgcn_wmma_f32_16x16x32_bf16(
                    false, aF, false, bF, (short)0, acc[t], false, false);
            }
        }

        // Fold this M-block into running argmin: d = c_sq[m] - 2 * cross.
#pragma unroll
        for (int t = 0; t < 8; ++t) {
            const int col = mBase + t * 16 + l16;
            const float cs = csq[col];
#pragma unroll
            for (int i = 0; i < 8; ++i) {
                const float d = __builtin_fmaf(-2.0f, acc[t][i], cs);
                if (d < best[i]) { best[i] = d; bidx[i] = col; }  // strict <: first-min
            }
        }
    }

    // Reduce across the 16 lanes of each half (tie -> lower index, jnp.argmin).
#pragma unroll
    for (int i = 0; i < 8; ++i) {
#pragma unroll
        for (int msk = 1; msk <= 8; msk <<= 1) {
            const float ob = __shfl_xor(best[i], msk, 32);
            const int   oi = __shfl_xor(bidx[i], msk, 32);
            if (ob < best[i] || (ob == best[i] && oi < bidx[i])) {
                best[i] = ob; bidx[i] = oi;
            }
        }
    }

    // Epilogue: this wave finalizes rows [rowBase, rowBase+16):
    // gather codebook[idx] -> outQ (exact f32), index -> outIdx.
    const int rowBase = blockRow + wave * 16;
#pragma unroll
    for (int r = 0; r < 16; ++r) {
        const int slot    = r & 7;
        const int srcLane = (r < 8) ? 0 : 16;
        const int idx     = __shfl(bidx[slot], srcLane, 32);

        const float4* crow = (const float4*)(cb + (size_t)idx * DIMS);
        float4*       orow = (float4*)(outQ + (size_t)(rowBase + r) * DIMS);
#pragma unroll
        for (int j = 0; j < 4; ++j) orow[lane + 32 * j] = crow[lane + 32 * j];
        if (lane == 0) outIdx[rowBase + r] = (float)idx;
    }
}

// ---------------------------------------------------------------------------
extern "C" void kernel_launch(void* const* d_in, const int* in_sizes, int n_in,
                              void* d_out, int out_size, void* d_ws, size_t ws_size,
                              hipStream_t stream) {
    (void)in_sizes; (void)n_in; (void)out_size; (void)ws_size;

    const float* emb = (const float*)d_in[0];   // [N, D] f32
    const float* cb  = (const float*)d_in[1];   // [M, D] f32

    float* outQ   = (float*)d_out;                      // [N, D] quantized
    float* outIdx = outQ + (size_t)N_NODES * DIMS;      // [N] indices (as f32)
    float* csq    = (float*)d_ws;                       // [M] ||c||^2

    csq_kernel<<<MCODE / 8, 256, 0, stream>>>(cb, csq);
    assign_kernel<<<N_NODES / BLOCK_ROWS, THREADS, 0, stream>>>(
        emb, cb, csq, outQ, outIdx);
}
